// BoxMaskHead_23811298689519
// MI455X (gfx1250) — matrix operand order repaired and checked by reading, not compiled, not run on hardware
//
#include <hip/hip_runtime.h>
#include <hip/hip_bf16.h>

// RoIAlign (jax.image.scale_and_translate, triangle kernel, antialias=True)
// One workgroup (256 threads = 8 waves) per box.
// Stage 1: y-contraction (<=33 taps) via float4 (b128) loads from L2-resident
//          feature map -> f16 LDS tile sRow[c][x] (padded stride).
// Stage 2: x-contraction as dense v_wmma_f32_16x16x32_f16, K-chunks of 32,
//          f32 accumulators persistent across chunks.

typedef __attribute__((ext_vector_type(16))) _Float16 v16h;
typedef __attribute__((ext_vector_type(8)))  float    v8f;

#define IN_H 512
#define IN_W 512
#define CHN  64
#define OUTW 32
#define CHUNK 32
#define ROWP  40        // padded row stride (halves): 80B keeps 16B alignment,
                        // spreads transposed b16 stores across banks
#define MAX_CHUNKS 16   // footprint <= 512 -> <= 16 chunks of 32
#define MAX_TAPS  36    // inv_scale <= 16 -> <= 33 taps

union U32H2 { unsigned u; _Float16 h[2]; };
union ABv   { v16h v; unsigned u[8]; };

__global__ __launch_bounds__(256)
void roialign_wmma_kernel(const float* __restrict__ fmap,
                          const float* __restrict__ boxes,
                          float* __restrict__ out)
{
  __shared__ _Float16 sWx[MAX_CHUNKS][OUTW][CHUNK]; // x-weights, packed-dense
  __shared__ _Float16 sRow[CHN][ROWP];              // stage-1 result [c][x]
  __shared__ float    sSfx[OUTW];                   // x sample centers
  __shared__ float    sInvNx[OUTW];                 // x renormalizers (0 if invalid)
  __shared__ float    sWy[MAX_TAPS];                // y-tap weights for current oy
  __shared__ int      sMeta[4];                     // xlo, nchunks, ylo, ky

  const int box = blockIdx.x;
  const int tid = threadIdx.x;

  // ---- box parameters (redundant per thread) ----
  const float cx = boxes[box*4+0], cy = boxes[box*4+1];
  float bw = boxes[box*4+2], bh = boxes[box*4+3];
  const float x0f = cx - bw*0.5f;          // computed before clamp, as in reference
  const float y0f = cy - bh*0.5f;
  bw = fmaxf(bw, 1e-6f); bh = fmaxf(bh, 1e-6f);
  const float invx = 16.0f*bw;             // 1/x_scale = bw*IN_W/OUTW
  const float invy = 16.0f*bh;
  const float ksx = fmaxf(invx, 1.0f);     // antialias kernel dilation
  const float ksy = fmaxf(invy, 1.0f);
  const float basex = 512.0f*x0f - 0.5f;   // sample_f(o) = (o+0.5)*inv + base
  const float basey = 512.0f*y0f - 0.5f;

  // ---- per-ox sample centers and renormalizers ----
  if (tid < OUTW) {
    const float s = ((float)tid + 0.5f)*invx + basex;
    sSfx[tid] = s;
    int lo = (int)floorf(s - ksx) + 1; if (lo < 0) lo = 0;
    int hi = (int)ceilf (s + ksx) - 1; if (hi > IN_W-1) hi = IN_W-1;
    float norm = 0.0f;
    for (int i = lo; i <= hi; ++i)
      norm += fmaxf(0.0f, 1.0f - fabsf(s - (float)i)/ksx);
    const bool valid = (s >= -0.5f) && (s <= (float)IN_W - 0.5f)
                       && (fabsf(norm) > 1.1920929e-4f);
    sInvNx[tid] = valid ? 1.0f/norm : 0.0f;
  }
  if (tid == 0) {
    const float s0  = 0.5f*invx + basex;
    const float s31 = 31.5f*invx + basex;
    int xlo = (int)floorf(s0  - ksx) + 1; if (xlo < 0) xlo = 0;
    int xhi = (int)ceilf (s31 + ksx) - 1; if (xhi > IN_W-1) xhi = IN_W-1;
    if (xhi < xlo) xhi = xlo;
    int nch = (xhi - xlo + 1 + CHUNK - 1)/CHUNK;
    if (nch > MAX_CHUNKS) nch = MAX_CHUNKS;
    sMeta[0] = xlo; sMeta[1] = nch;
  }
  __syncthreads();
  const int xlo = sMeta[0];
  const int nch = sMeta[1];

  // ---- build packed-dense f16 Wx for all chunks (oy-invariant) ----
  for (int idx = tid; idx < nch*OUTW*CHUNK; idx += 256) {
    const int xi = idx % CHUNK;
    const int ox = (idx / CHUNK) % OUTW;
    const int ch = idx / (CHUNK*OUTW);
    const int x  = xlo + ch*CHUNK + xi;
    float w = 0.0f;
    if (x < IN_W) {
      const float s = sSfx[ox];
      w = fmaxf(0.0f, 1.0f - fabsf(s - (float)x)/ksx) * sInvNx[ox];
    }
    sWx[ch][ox][xi] = (_Float16)w;
  }

  // ---- role decomposition ----
  const int lane  = tid & 31;
  const int wave  = tid >> 5;         // 8 waves
  const int mtile = wave & 1;         // ox tile 0..1
  const int ctile = wave >> 1;        // c  tile 0..3
  const int nIdx  = lane & 15;        // N within WMMA tile
  const int hHalf = lane >> 4;        // lane half
  // stage-1 mapping: 16 float4 channel-groups x 32 x-positions = 512 slots,
  // thread owns (cg, xi0) and (cg, xi0+16)
  const int cg  = tid & 15;           // c = 4*cg .. 4*cg+3
  const int xi0 = tid >> 4;           // 0..15
  const int xi1 = xi0 + 16;

  for (int oy = 0; oy < OUTW; ++oy) {
    // y-tap weights (thread 0, <=33 scalar iterations)
    if (tid == 0) {
      const float s = ((float)oy + 0.5f)*invy + basey;
      int lo = (int)floorf(s - ksy) + 1; if (lo < 0) lo = 0;
      int hi = (int)ceilf (s + ksy) - 1; if (hi > IN_H-1) hi = IN_H-1;
      float norm = 0.0f;
      for (int i = lo; i <= hi; ++i)
        norm += fmaxf(0.0f, 1.0f - fabsf(s - (float)i)/ksy);
      const bool valid = (hi >= lo) && (s >= -0.5f) && (s <= (float)IN_H - 0.5f)
                         && (fabsf(norm) > 1.1920929e-4f);
      int ky = valid ? (hi - lo + 1) : 0;
      if (ky > MAX_TAPS) ky = MAX_TAPS;
      const float innorm = valid ? 1.0f/norm : 0.0f;
      for (int t = 0; t < ky; ++t)
        sWy[t] = fmaxf(0.0f, 1.0f - fabsf(s - (float)(lo+t))/ksy) * innorm;
      sMeta[2] = lo; sMeta[3] = ky;
    }
    __syncthreads();
    const int ylo = sMeta[2];
    const int ky  = sMeta[3];

    v8f acc = {}; // persistent 16x16 f32 accumulator across K-chunks

    for (int ch = 0; ch < nch; ++ch) {
      // ---- stage 1: rowacc[c][x] = sum_t wy[t] * F[ylo+t][x][c] ----
      // float4 loads; clamped 32-bit offsets hoisted out of the tap loop
      int xg0 = xlo + ch*CHUNK + xi0; if (xg0 > IN_W-1) xg0 = IN_W-1;
      int xg1 = xlo + ch*CHUNK + xi1; if (xg1 > IN_W-1) xg1 = IN_W-1;
      const int off0 = xg0*(CHN/4) + cg;        // float4 units within a row
      const int off1 = xg1*(CHN/4) + cg;
      float4 a0 = make_float4(0.f,0.f,0.f,0.f);
      float4 a1 = make_float4(0.f,0.f,0.f,0.f);
      const float4* p4 = (const float4*)fmap + (size_t)ylo*(IN_W*CHN/4);
      for (int t = 0; t < ky; ++t) {
        const float wv = sWy[t];
        const float4 f0 = p4[off0];
        const float4 f1 = p4[off1];
        a0.x = fmaf(wv, f0.x, a0.x);  a0.y = fmaf(wv, f0.y, a0.y);
        a0.z = fmaf(wv, f0.z, a0.z);  a0.w = fmaf(wv, f0.w, a0.w);
        a1.x = fmaf(wv, f1.x, a1.x);  a1.y = fmaf(wv, f1.y, a1.y);
        a1.z = fmaf(wv, f1.z, a1.z);  a1.w = fmaf(wv, f1.w, a1.w);
        p4 += IN_W*CHN/4;
      }
      __syncthreads(); // previous chunk's B-operand reads complete
      {
        const int c0 = cg*4;
        sRow[c0+0][xi0] = (_Float16)a0.x;  sRow[c0+0][xi1] = (_Float16)a1.x;
        sRow[c0+1][xi0] = (_Float16)a0.y;  sRow[c0+1][xi1] = (_Float16)a1.y;
        sRow[c0+2][xi0] = (_Float16)a0.z;  sRow[c0+2][xi1] = (_Float16)a1.z;
        sRow[c0+3][xi0] = (_Float16)a0.w;  sRow[c0+3][xi1] = (_Float16)a1.w;
      }
      __syncthreads(); // rowacc visible to all waves

      // ---- stage 2: one WMMA per wave, K = 32 ----
      ABv A, B;
      #pragma unroll
      for (int v = 0; v < 8; ++v) {
        // A 16x32 f16 layout: lane M=lane%16; V0-3: K = h*8 + 2v(+1); V4-7: +16
        const int kb = ((v < 4) ? 0 : 16) + hHalf*8 + (v & 3)*2;
        A.u[v] = *(const unsigned*)&sWx[ch][mtile*16 + nIdx][kb];
        // B 32x16 f16 layout: lane N=lane%16; V0-3: K = h*8 + 2v(+1); V4-7: +16
        B.u[v] = *(const unsigned*)&sRow[ctile*16 + nIdx][kb];
      }
      acc = __builtin_amdgcn_wmma_f32_16x16x32_f16(
              /*neg_a=*/false, A.v, /*neg_b=*/false, B.v,
              /*c_mod=*/(short)0, acc, /*reuse_a=*/false, /*reuse_b=*/false);
    }

    // ---- store D: VGPR r -> M = r + h*8 (ox), N (c) ----
    const int cglob = ctile*16 + nIdx;
    float* op = out + (((size_t)box*OUTW + oy)*OUTW)*CHN + cglob;
    #pragma unroll
    for (int rr = 0; rr < 8; ++rr) {
      const int ox = mtile*16 + rr + hHalf*8;
      op[(size_t)ox*CHN] = acc[rr];
    }
  }
}

extern "C" void kernel_launch(void* const* d_in, const int* in_sizes, int n_in,
                              void* d_out, int out_size, void* d_ws, size_t ws_size,
                              hipStream_t stream) {
  const float* fmap  = (const float*)d_in[0];
  const float* boxes = (const float*)d_in[1];
  float* out = (float*)d_out;
  const int nbox = in_sizes[1] / 4;
  roialign_wmma_kernel<<<nbox, 256, 0, stream>>>(fmap, boxes, out);
}
